// VQVAE_87179246174671
// MI455X (gfx1250) — matrix operand order repaired and checked by reading, not compiled, not run on hardware
//
#include <hip/hip_runtime.h>
#include <hip/hip_bf16.h>

typedef __attribute__((ext_vector_type(16))) _Float16 v16h;
typedef __attribute__((ext_vector_type(8)))  float    v8f;
typedef unsigned short u16;

#define WMMA_F16(A, B, C, RA) \
  __builtin_amdgcn_wmma_f32_16x16x32_f16(false, (A), false, (B), (short)0, (C), (RA), false)

__device__ __forceinline__ u16 h2u(_Float16 h) {
  u16 v; __builtin_memcpy(&v, &h, 2); return v;
}

// Assemble a 16-half WMMA fragment from two 16-byte LDS chunks.
__device__ __forceinline__ v16h lds_frag(const u16* p0, const u16* p1) {
  v16h r;
  uint4* rp = reinterpret_cast<uint4*>(&r);
  rp[0] = *reinterpret_cast<const uint4*>(p0);
  rp[1] = *reinterpret_cast<const uint4*>(p1);
  return r;
}

// ---------------------------------------------------------------------------
// Unified implicit-GEMM conv / transposed-conv (stride-2) kernel.
//   NN   : Cout (32|64), KCH : number of 32-wide K chunks (= KH*KW*Cin/32)
//   CINC : Cin/32, MODE : 0 = fwd conv (pad_low 0), 1 = conv_transpose s=2
//   RELU : apply ReLU in epilogue
// Block: 256 threads (8 waves). M-tile = 128 output pixels, full N resident.
// A: double-buffered LDS tiles (software pipeline, 1 barrier / k-chunk).
// B: whole filter in LDS once, fragment-ready layout [kc][n][k] (80B pitch).
// ---------------------------------------------------------------------------
template<int NN, int KCH, int CINC, int MODE, int RELU>
__global__ __launch_bounds__(256) void conv_gemm(
    const _Float16* __restrict__ in, _Float16* __restrict__ out,
    const float* __restrict__ w, const float* __restrict__ bias,
    int Hin, int Win, int Hout, int Wout, int KW, int stride)
{
  constexpr int NT   = NN / 16;
  constexpr int TAPS = KCH / CINC;
  constexpr int Cin  = CINC * 32;
  extern __shared__ char smem[];
  u16* Abuf[2] = { (u16*)smem, (u16*)(smem + 10240) };   // 2 x 128*40 halves
  u16* Blds    = (u16*)(smem + 20480);                   // KCH * NN * 40 halves

  const int t    = threadIdx.x;
  const int lane = t & 31;
  const int wv   = t >> 5;
  const int block_row = blockIdx.x * 128;
  const int hw   = Hout * Wout;

  // ---- one-time filter load into fragment-ready LDS layout (coalesced on n)
  for (int e = t; e < KCH * 32 * NN; e += 256) {
    int kc = e / (32 * NN);
    int r2 = e - kc * (32 * NN);
    int j  = r2 / NN;
    int n  = r2 - j * NN;
    Blds[(kc * NN + n) * 40 + j] = h2u((_Float16)w[(kc * 32 + j) * NN + n]);
  }

  // ---- per-thread output-pixel decode (2 threads per A row)
  const int r    = t >> 1;
  const int hs   = t & 1;
  const int rowg = block_row + r;
  const int bb   = rowg / hw;
  const int rem  = rowg - bb * hw;
  const int oy   = rem / Wout;
  const int ox   = rem - oy * Wout;

  // ---- precompute per-tap source pointers + validity (constant-folds taps)
  const _Float16* tp[TAPS];
  bool tv[TAPS];
#pragma unroll
  for (int tap = 0; tap < TAPS; ++tap) {
    int ky = tap / KW, kx = tap - ky * KW;
    int iy, ix; bool valid;
    if (MODE == 0) {
      iy = oy * stride + ky; ix = ox * stride + kx;
      valid = (iy < Hin) && (ix < Win);
    } else {
      int ty = oy + 1 - ky, tx = ox + 1 - kx;
      valid = (ty >= 0) && (tx >= 0) && !(ty & 1) && !(tx & 1);
      iy = ty >> 1; ix = tx >> 1;
      valid = valid && (iy < Hin) && (ix < Win);
    }
    tv[tap] = valid;
    tp[tap] = in + (((long)bb * Hin + iy) * Win + ix) * Cin + hs * 16;
  }

  auto gather = [&](int kc, u16* Ab) {
    int tap = kc / CINC;
    int cis = (kc - tap * CINC) << 5;
    uint4 u0 = make_uint4(0u, 0u, 0u, 0u), u1 = u0;
    if (tv[tap]) {
      const uint4* g = reinterpret_cast<const uint4*>(tp[tap] + cis);
      u0 = g[0]; u1 = g[1];
    }
    *reinterpret_cast<uint4*>(Ab + r * 40 + hs * 16)     = u0;
    *reinterpret_cast<uint4*>(Ab + r * 40 + hs * 16 + 8) = u1;
  };

  v8f acc[NT];
  v8f zero8 = {0.f, 0.f, 0.f, 0.f, 0.f, 0.f, 0.f, 0.f};
#pragma unroll
  for (int i = 0; i < NT; ++i) acc[i] = zero8;

  gather(0, Abuf[0]);
  __syncthreads();                    // B + first A tile resident

  const int arow0 = wv * 16 + (lane & 15);
  const int kb    = (lane >> 4) << 3;
  const int jb    = (lane >> 4) << 3;

#pragma unroll
  for (int kc = 0; kc < KCH; ++kc) {
    u16* cur = Abuf[kc & 1];
    u16* nxt = Abuf[(kc & 1) ^ 1];
    if (kc + 1 < KCH) gather(kc + 1, nxt);          // pipelined gather
    if (kc + 2 < KCH) {                             // prefetch chunk after next
      int tap2 = (kc + 2) / CINC;
      int cis2 = ((kc + 2) - tap2 * CINC) << 5;
      if (tv[tap2]) __builtin_prefetch(tp[tap2] + cis2, 0, 1);
    }

    v16h a = lds_frag(cur + arow0 * 40 + kb, cur + arow0 * 40 + kb + 16);
    auto bfrag = [&](int nt2) {
      const u16* bp = Blds + (kc * NN + nt2 * 16 + (lane & 15)) * 40 + jb;
      return lds_frag(bp, bp + 16);
    };
    // reuse-A hint (literal constants) on the identical repeat instructions
    acc[0] = WMMA_F16(a, bfrag(0), acc[0], false);
    acc[1] = WMMA_F16(a, bfrag(1), acc[1], true);
    if constexpr (NT > 2) {
      acc[2] = WMMA_F16(a, bfrag(2), acc[2], true);
      acc[3] = WMMA_F16(a, bfrag(3), acc[3], true);
    }
    __syncthreads();
  }

  // ---- epilogue: bias (+ReLU), f16 store (NHWC)
#pragma unroll
  for (int nt2 = 0; nt2 < NT; ++nt2) {
    int ncol = nt2 * 16 + (lane & 15);
    float bv = bias[ncol];
#pragma unroll
    for (int v = 0; v < 8; ++v) {
      int mrow = wv * 16 + ((lane >> 4) << 3) + v;
      float val = acc[nt2][v] + bv;
      if (RELU) val = fmaxf(val, 0.f);
      out[(long)(block_row + mrow) * NN + ncol] = (_Float16)val;
    }
  }
}

// ---------------------------------------------------------------------------
// Vector quantization: argmin_k ||e_k||^2 - 2 z.e_k via WMMA score GEMM,
// then gather quantized rows from the f32 embedding table.
// Block: 256 threads (8 waves), 16 z-rows per block, 512 codes in 4 chunks.
// ---------------------------------------------------------------------------
__global__ __launch_bounds__(256) void vq_kernel(
    const _Float16* __restrict__ z, const float* __restrict__ emb,
    const float* __restrict__ enorm, _Float16* __restrict__ q)
{
  extern __shared__ char smem[];
  u16*   Alds   = (u16*)smem;                            // 16 * 72 halves
  u16*   Elds   = (u16*)(smem + 2304);                   // 2 * 128 * 40 halves
  float* scores = (float*)(smem + 2304 + 20480);         // 16 * 520 f32
  float* enl    = (float*)(smem + 2304 + 20480 + 33280); // 512 f32
  float* rv     = enl + 512;                             // 256 f32
  int*   ri     = (int*)(rv + 256);                      // 256 i32
  int*   idxL   = ri + 256;                              // 16 i32

  const int t = threadIdx.x, lane = t & 31, wv = t >> 5;
  const int row_base = blockIdx.x * 16;

  for (int i = t; i < 512; i += 256) enl[i] = enorm[i];
  if (t < 64) {
    int row = t >> 2, seg = t & 3;
    const uint4* g = reinterpret_cast<const uint4*>(
        z + (long)(row_base + row) * 64 + seg * 16);
    *reinterpret_cast<uint4*>(Alds + row * 72 + seg * 16)     = g[0];
    *reinterpret_cast<uint4*>(Alds + row * 72 + seg * 16 + 8) = g[1];
  }
  __syncthreads();

  int arow = lane & 15, kb = (lane >> 4) << 3;
  v16h a0 = lds_frag(Alds + arow * 72 + kb,      Alds + arow * 72 + kb + 16);
  v16h a1 = lds_frag(Alds + arow * 72 + 32 + kb, Alds + arow * 72 + 32 + kb + 16);

  for (int nc = 0; nc < 4; ++nc) {
    if (nc) __syncthreads();
    for (int e = t; e < 8192; e += 256) {       // E chunk -> fragment layout
      int kc = e >> 12, r2 = e & 4095, j = r2 >> 7, n = r2 & 127;
      Elds[(kc * 128 + n) * 40 + j] =
          h2u((_Float16)emb[(kc * 32 + j) * 512 + nc * 128 + n]);
    }
    __syncthreads();

    int c0 = wv * 16;
    int jb = (lane >> 4) << 3;
    const u16* b0p = Elds + (c0 + (lane & 15)) * 40 + jb;
    const u16* b1p = Elds + (128 + c0 + (lane & 15)) * 40 + jb;
    v8f acc = {0.f, 0.f, 0.f, 0.f, 0.f, 0.f, 0.f, 0.f};
    acc = WMMA_F16(a0, lds_frag(b0p, b0p + 16), acc, false);
    acc = WMMA_F16(a1, lds_frag(b1p, b1p + 16), acc, false);

    int ncol = nc * 128 + c0 + (lane & 15);
    float en = enl[ncol];
#pragma unroll
    for (int v = 0; v < 8; ++v) {
      int m = ((lane >> 4) << 3) + v;
      scores[m * 520 + ncol] = en - 2.0f * acc[v];
    }
  }
  __syncthreads();

  {  // per-(row, 32-col-slice) argmin
    int row = t >> 4, g = t & 15;
    float best = 3.0e38f; int bi = 0;
    for (int c = g * 32; c < g * 32 + 32; ++c) {
      float s = scores[row * 520 + c];
      if (s < best) { best = s; bi = c; }
    }
    rv[t] = best; ri[t] = bi;
  }
  __syncthreads();
  if (t < 16) {
    float best = 3.0e38f; int bi = 0;
    for (int g = 0; g < 16; ++g) {
      float s = rv[t * 16 + g];
      if (s < best) { best = s; bi = ri[t * 16 + g]; }
    }
    idxL[t] = bi;
  }
  __syncthreads();

  {  // straight-through forward output == quantized row gather
    int row = t >> 4, ds = t & 15;
    int k = idxL[row];
#pragma unroll
    for (int dd = 0; dd < 4; ++dd) {
      int d = ds * 4 + dd;
      q[(long)(row_base + row) * 64 + d] = (_Float16)emb[d * 512 + k];
    }
  }
}

// ---------------------------------------------------------------------------
// enc conv1: 3x3 stride-2, Cin=1 -> Cout=32, ReLU. One thread per out pixel.
// ---------------------------------------------------------------------------
__global__ __launch_bounds__(256) void conv1_kernel(
    const float* __restrict__ x, const float* __restrict__ w,
    const float* __restrict__ bias, _Float16* __restrict__ out)
{
  __shared__ float wl[288];
  __shared__ float bl[32];
  int t = threadIdx.x;
  if (t < 288) wl[t] = w[t];
  if (t < 32)  bl[t] = bias[t];
  __syncthreads();

  int gid = blockIdx.x * 256 + t;          // 32*128*128
  int bb = gid >> 14, rem = gid & 16383;
  int oy = rem >> 7,  ox = rem & 127;

  float acc[32];
#pragma unroll
  for (int c = 0; c < 32; ++c) acc[c] = 0.f;
#pragma unroll
  for (int ky = 0; ky < 3; ++ky) {
    int iy = oy * 2 + ky;
    if (iy >= 256) continue;
#pragma unroll
    for (int kx = 0; kx < 3; ++kx) {
      int ix = ox * 2 + kx;
      if (ix >= 256) continue;
      float xv = x[((long)bb * 256 + iy) * 256 + ix];
#pragma unroll
      for (int c = 0; c < 32; ++c)
        acc[c] = fmaf(xv, wl[(ky * 3 + kx) * 32 + c], acc[c]);
    }
  }
#pragma unroll
  for (int c = 0; c < 32; ++c)
    out[(long)gid * 32 + c] = (_Float16)fmaxf(acc[c] + bl[c], 0.f);
}

// ---------------------------------------------------------------------------
// dec deconv3: stride-1 3x3 conv_transpose, 32 -> 1 channel, f32 output.
// ---------------------------------------------------------------------------
__global__ __launch_bounds__(256) void dec3_kernel(
    const _Float16* __restrict__ h, const float* __restrict__ w,
    const float* __restrict__ bias, float* __restrict__ out)
{
  __shared__ float wl[288];
  int t = threadIdx.x;
  if (t < 288) wl[t] = w[t];
  __syncthreads();

  int gid = blockIdx.x * 256 + t;          // 32*256*256
  int bb = gid >> 16, rem = gid & 65535;
  int oy = rem >> 8,  ox = rem & 255;

  float acc = bias[0];
  for (int ky = 0; ky < 3; ++ky) {
    int iy = oy + 1 - ky;
    if (iy < 0 || iy >= 256) continue;
    for (int kx = 0; kx < 3; ++kx) {
      int ix = ox + 1 - kx;
      if (ix < 0 || ix >= 256) continue;
      const _Float16* p = h + (((long)bb * 256 + iy) * 256 + ix) * 32;
#pragma unroll
      for (int c = 0; c < 32; ++c)
        acc = fmaf((float)p[c], wl[(ky * 3 + kx) * 32 + c], acc);
    }
  }
  out[gid] = acc;
}

// ---------------------------------------------------------------------------
// ||e_k||^2 for each of the 512 codes.
// ---------------------------------------------------------------------------
__global__ void enorm_kernel(const float* __restrict__ E, float* __restrict__ en)
{
  int c = blockIdx.x * blockDim.x + threadIdx.x;
  if (c < 512) {
    float s = 0.f;
    for (int d = 0; d < 64; ++d) { float v = E[d * 512 + c]; s = fmaf(v, v, s); }
    en[c] = s;
  }
}

// ---------------------------------------------------------------------------
extern "C" void kernel_launch(void* const* d_in, const int* in_sizes, int n_in,
                              void* d_out, int out_size, void* d_ws, size_t ws_size,
                              hipStream_t stream)
{
  (void)in_sizes; (void)n_in; (void)out_size; (void)ws_size;
  const float* x   = (const float*)d_in[0];
  const float* ew1 = (const float*)d_in[1];
  const float* eb1 = (const float*)d_in[2];
  const float* ew2 = (const float*)d_in[3];
  const float* eb2 = (const float*)d_in[4];
  const float* ew3 = (const float*)d_in[5];
  const float* eb3 = (const float*)d_in[6];
  const float* emb = (const float*)d_in[7];
  const float* dw1 = (const float*)d_in[8];
  const float* db1 = (const float*)d_in[9];
  const float* dw2 = (const float*)d_in[10];
  const float* db2 = (const float*)d_in[11];
  const float* dw3 = (const float*)d_in[12];
  const float* db3 = (const float*)d_in[13];

  char* ws = (char*)d_ws;
  size_t off = 0;
  auto alloc = [&](size_t bytes) {
    void* p = ws + off; off += (bytes + 255) & ~(size_t)255; return p;
  };
  _Float16* h1 = (_Float16*)alloc((size_t)32 * 128 * 128 * 32 * 2);
  _Float16* h2 = (_Float16*)alloc((size_t)32 * 64 * 64 * 64 * 2);
  _Float16* zb = (_Float16*)alloc((size_t)131072 * 64 * 2);
  _Float16* qb = (_Float16*)alloc((size_t)131072 * 64 * 2);
  _Float16* d1 = (_Float16*)alloc((size_t)524288 * 64 * 2);
  _Float16* d2 = (_Float16*)alloc((size_t)2097152 * 32 * 2);
  float*    en = (float*)alloc(512 * 4);

  enorm_kernel<<<dim3(2), dim3(256), 0, stream>>>(emb, en);
  conv1_kernel<<<dim3(2048), dim3(256), 0, stream>>>(x, ew1, eb1, h1);

  // enc conv2: [32,128,128,32] -> [32,64,64,64], 3x3 s2, ReLU
  conv_gemm<64, 9, 1, 0, 1><<<dim3(1024), dim3(256),
      (size_t)(20480 + 9 * 64 * 80), stream>>>(
      h1, h2, ew2, eb2, 128, 128, 64, 64, 3, 2);

  // enc conv3: 1x1 s1 -> z [32,64,64,64]
  conv_gemm<64, 2, 2, 0, 0><<<dim3(1024), dim3(256),
      (size_t)(20480 + 2 * 64 * 80), stream>>>(
      h2, zb, ew3, eb3, 64, 64, 64, 64, 1, 1);

  // vector quantization -> q
  vq_kernel<<<dim3(8192), dim3(256),
      (size_t)(2304 + 20480 + 33280 + 2048 + 1024 + 1024 + 64), stream>>>(
      zb, emb, en, qb);

  // dec deconv1: [32,64,64,64] -> [32,128,128,64], 3x3 s2 transpose, ReLU
  conv_gemm<64, 18, 2, 1, 1><<<dim3(4096), dim3(256),
      (size_t)(20480 + 18 * 64 * 80), stream>>>(
      qb, d1, dw1, db1, 64, 64, 128, 128, 3, 2);

  // dec deconv2: [32,128,128,64] -> [32,256,256,32], 3x3 s2 transpose, ReLU
  conv_gemm<32, 18, 2, 1, 1><<<dim3(16384), dim3(256),
      (size_t)(20480 + 18 * 32 * 80), stream>>>(
      d1, d2, dw2, db2, 128, 128, 256, 256, 3, 2);

  // dec deconv3: [32,256,256,32] -> [32,256,256,1] f32
  dec3_kernel<<<dim3(8192), dim3(256), 0, stream>>>(d2, dw3, db3, (float*)d_out);
}
